// Block_70712341562194
// MI455X (gfx1250) — compile-verified, matrix operand order checked by reading, not run on hardware
//
#include <hip/hip_runtime.h>
#include <hip/hip_bf16.h>

// ---------------------------------------------------------------------------
// Transformer block (pre-LN, MHA + MLP) for MI455X / gfx1250.
// GEMMs: v_wmma_f32_16x16x32_bf16 (wave32, 16x16 tiles, K=32), fed by
// CDNA5 async global->LDS copies (ASYNCcnt) with LDS ping-pong buffering.
// ---------------------------------------------------------------------------

typedef __attribute__((ext_vector_type(16))) __bf16 v16bf;
typedef __attribute__((ext_vector_type(8)))  float  v8f;

struct Pack32 { uint4 lo; uint4 hi; };   // 32 bytes == v16bf

__device__ __forceinline__ __bf16 f2bf(float f) {
  unsigned u = __builtin_bit_cast(unsigned, f);
  unsigned r = u + 0x7FFFu + ((u >> 16) & 1u);   // round-to-nearest-even
  unsigned short s = (unsigned short)(r >> 16);
  return __builtin_bit_cast(__bf16, s);
}

// Low 32 bits of a generic pointer to __shared__ == wave-relative LDS byte
// address (flat aperture keeps the LDS offset in addr[31:0]).
__device__ __forceinline__ unsigned lds_addr(const void* p) {
  return (unsigned)(unsigned long long)(uintptr_t)p;
}

// CDNA5 async copy: LDS[vdst_lane] = MEM[vaddr_lane], 16 bytes, ASYNCcnt.
__device__ __forceinline__ void async_load_b128(unsigned lds, const void* gptr) {
  asm volatile("global_load_async_to_lds_b128 %0, %1, off"
               :: "v"(lds), "v"((unsigned long long)(uintptr_t)gptr)
               : "memory");
}

constexpr int BSZ   = 4;
constexpr int NSEQ  = 1024;
constexpr int CD    = 1024;
constexpr int HEADS = 16;
constexpr int HDID  = 64;
constexpr int HID   = 4096;
constexpr int MROWS = BSZ * NSEQ;        // 4096 flattened (b, n) rows
constexpr float LN_EPS = 1e-5f;

// ---------------------------------------------------------------------------
// Generic TN WMMA GEMM:  out[m, n] = sum_k A[m, k] * W[n, k]   (+ epilogue)
// MODE: 0=QKV(bf16 out, q scaled)  1=SCORES(f32 out, per-(b,h))
//       2=AV(A is f32 attn, BN=64, bf16 out scatter)  3=PROJ(resid+gamma)
//       4=FC1(GELU, bf16 out)      5=FC2(resid+gamma, f32 out)
// ---------------------------------------------------------------------------
template <int MODE>
__global__ __launch_bounds__(256) void gemm_kernel(
    const void* __restrict__ Ain, const __bf16* __restrict__ Win,
    void* __restrict__ Out,
    const float* __restrict__ bias, const float* __restrict__ resid,
    const float* __restrict__ gamma)
{
  constexpr bool AF32  = (MODE == 2);
  constexpr bool ASYNC = !AF32;                  // async-to-LDS staging
  constexpr int BM = 128;
  constexpr int BN = (MODE == 2) ? 64 : 128;
  constexpr int K   = (MODE == 1) ? 64   : (MODE == 5 ? 4096 : 1024);
  constexpr int LDA = (MODE == 1) ? 3072 : (MODE == 5 ? 4096 : 1024);
  constexpr int LDW = LDA;                       // holds for every mode used
  constexpr int LDO = (MODE == 0) ? 3072 : (MODE == 4 ? 4096 : 1024);
  constexpr int KS    = 32;
  constexpr int NSTEP = K / KS;
  constexpr int PITCH = 48;                      // bf16 elems; 96B, 16B-aligned
  constexpr int WM = 32;                         // 4 waves along M
  constexpr int WN = BN / 2;                     // 2 waves along N
  constexpr int MT = WM / 16;                    // 2 tiles
  constexpr int NT = WN / 16;                    // 4 (or 2 for AV)

  __shared__ __align__(16) __bf16 As[2][BM][PITCH];
  __shared__ __align__(16) __bf16 Ws[2][BN][PITCH];

  const int t     = threadIdx.x;
  const int lane  = t & 31;
  const int wid   = t >> 5;
  const int waveM = wid & 3;
  const int waveN = wid >> 2;
  const int z     = blockIdx.z;

  size_t aOff = 0, wOff = 0, oOff = 0;
  if constexpr (MODE == 1) {                     // S = q_bh * k_bh^T
    size_t bh = (size_t)(z >> 4) * NSEQ * 3072;
    aOff = bh + (size_t)(z & 15) * HDID;                 // q slice
    wOff = bh + CD + (size_t)(z & 15) * HDID;            // k slice
    oOff = (size_t)z * NSEQ * NSEQ;                      // attn[b,h]
  } else if constexpr (MODE == 2) {              // AO = attn * v
    aOff = (size_t)z * NSEQ * NSEQ;                      // attn[b,h] (f32)
    wOff = (size_t)z * HDID * NSEQ;                      // vT[b,h]
    oOff = (size_t)(z >> 4) * NSEQ * CD + (size_t)(z & 15) * HDID; // ao scatter
  }
  const __bf16* Abf = (const __bf16*)Ain + (AF32 ? 0 : aOff);
  const float*  Af  = (const float*)Ain + aOff;
  const __bf16* W   = Win + wOff;

  const int rowBase = blockIdx.y * BM;
  const int colBase = blockIdx.x * BN;

  v8f acc[MT][NT];
  for (int i = 0; i < MT; ++i)
    for (int j = 0; j < NT; ++j)
      for (int e = 0; e < 8; ++e) acc[i][j][e] = 0.0f;

  // Per-thread staging coordinates (2 threads per row, 16 bf16 each).
  const int ar = t >> 1;
  const int ac = (t & 1) * 16;

  const int lrow = lane & 15;
  const int kh   = (lane >> 4) * 8;

  if constexpr (ASYNC) {
    // Issue one K-tile (A: 128x32, W: 128x32) into LDS buffer `buf`.
    // 4 async b128 instructions per wave per tile.
    auto issue_tile = [&](int step, int buf) {
      const __bf16* ga = Abf + (size_t)(rowBase + ar) * LDA + step * KS + ac;
      const unsigned la = lds_addr(&As[buf][ar][ac]);
      async_load_b128(la, ga);
      async_load_b128(la + 16u, ga + 8);
      const __bf16* gw = W + (size_t)(colBase + ar) * LDW + step * KS + ac;
      const unsigned lw = lds_addr(&Ws[buf][ar][ac]);
      async_load_b128(lw, gw);
      async_load_b128(lw + 16u, gw + 8);
    };

    issue_tile(0, 0);
    for (int step = 0; step < NSTEP; ++step) {
      const int cur = step & 1;
      if (step + 1 < NSTEP) {
        issue_tile(step + 1, cur ^ 1);
        // current tile done when only the 4 just-issued remain outstanding
        asm volatile("s_wait_asynccnt 0x4" ::: "memory");
      } else {
        asm volatile("s_wait_asynccnt 0x0" ::: "memory");
      }
      __syncthreads();

      v16bf afr[MT];
      for (int mt = 0; mt < MT; ++mt) {
        const int r = waveM * WM + mt * 16 + lrow;
        Pack32 p;
        p.lo = *(const uint4*)&As[cur][r][kh];
        p.hi = *(const uint4*)&As[cur][r][16 + kh];
        afr[mt] = __builtin_bit_cast(v16bf, p);
      }
      v16bf wfr[NT];
      for (int nt = 0; nt < NT; ++nt) {
        const int r = waveN * WN + nt * 16 + lrow;
        Pack32 p;
        p.lo = *(const uint4*)&Ws[cur][r][kh];
        p.hi = *(const uint4*)&Ws[cur][r][16 + kh];
        wfr[nt] = __builtin_bit_cast(v16bf, p);
      }
      for (int mt = 0; mt < MT; ++mt)
        for (int nt = 0; nt < NT; ++nt)
          acc[mt][nt] = __builtin_amdgcn_wmma_f32_16x16x32_bf16(
              false, afr[mt], false, wfr[nt], (short)0, acc[mt][nt], false, false);
      __syncthreads();   // all reads of buf `cur` done before it is reissued
    }
  } else {
    // Synchronous path (A is f32 attn: convert to bf16 while staging).
    for (int kk = 0; kk < K; kk += KS) {
      __syncthreads();
      {
        const float4* src = (const float4*)(Af + (size_t)(rowBase + ar) * LDA + kk + ac);
        float4 f0 = src[0], f1 = src[1], f2 = src[2], f3 = src[3];
        __bf16 tb[16] __attribute__((aligned(16)));
        tb[0]=f2bf(f0.x); tb[1]=f2bf(f0.y); tb[2]=f2bf(f0.z); tb[3]=f2bf(f0.w);
        tb[4]=f2bf(f1.x); tb[5]=f2bf(f1.y); tb[6]=f2bf(f1.z); tb[7]=f2bf(f1.w);
        tb[8]=f2bf(f2.x); tb[9]=f2bf(f2.y); tb[10]=f2bf(f2.z); tb[11]=f2bf(f2.w);
        tb[12]=f2bf(f3.x); tb[13]=f2bf(f3.y); tb[14]=f2bf(f3.z); tb[15]=f2bf(f3.w);
        *(uint4*)&As[0][ar][ac]     = *(uint4*)&tb[0];
        *(uint4*)&As[0][ar][ac + 8] = *(uint4*)&tb[8];
      }
      if (ar < BN) {
        const uint4* src = (const uint4*)(W + (size_t)(colBase + ar) * LDW + kk + ac);
        uint4 u0 = src[0];
        uint4 u1 = src[1];
        *(uint4*)&Ws[0][ar][ac]     = u0;
        *(uint4*)&Ws[0][ar][ac + 8] = u1;
      }
      __syncthreads();

      v16bf afr[MT];
      for (int mt = 0; mt < MT; ++mt) {
        const int r = waveM * WM + mt * 16 + lrow;
        Pack32 p;
        p.lo = *(const uint4*)&As[0][r][kh];
        p.hi = *(const uint4*)&As[0][r][16 + kh];
        afr[mt] = __builtin_bit_cast(v16bf, p);
      }
      v16bf wfr[NT];
      for (int nt = 0; nt < NT; ++nt) {
        const int r = waveN * WN + nt * 16 + lrow;
        Pack32 p;
        p.lo = *(const uint4*)&Ws[0][r][kh];
        p.hi = *(const uint4*)&Ws[0][r][16 + kh];
        wfr[nt] = __builtin_bit_cast(v16bf, p);
      }
      for (int mt = 0; mt < MT; ++mt)
        for (int nt = 0; nt < NT; ++nt)
          acc[mt][nt] = __builtin_amdgcn_wmma_f32_16x16x32_bf16(
              false, afr[mt], false, wfr[nt], (short)0, acc[mt][nt], false, false);
    }
  }

  // Epilogue. C layout: VGPR r -> M = r + 8*(lane>=16); N = lane&15.
  const int lrow8 = (lane >> 4) * 8;
  const int lcol  = lane & 15;
  // q-scale is block-uniform for MODE 0 (BN=128 divides C).
  const float qscale = (colBase < CD) ? 0.125f : 1.0f;   // HD^-0.5
  for (int mt = 0; mt < MT; ++mt) {
    for (int nt = 0; nt < NT; ++nt) {
      const int gr0 = rowBase + waveM * WM + mt * 16 + lrow8;
      const int gc  = colBase + waveN * WN + nt * 16 + lcol;
      for (int r = 0; r < 8; ++r) {
        const int gr = gr0 + r;
        float v = acc[mt][nt][r];
        const size_t oi = oOff + (size_t)gr * LDO + gc;
        if constexpr (MODE == 0) {               // qkv: scale q columns
          ((__bf16*)Out)[oi] = f2bf(v * qscale);
        } else if constexpr (MODE == 1) {        // raw scores (f32)
          ((float*)Out)[oi] = v;
        } else if constexpr (MODE == 2) {        // attention output (bf16)
          ((__bf16*)Out)[oi] = f2bf(v);
        } else if constexpr (MODE == 3) {        // proj + gamma1 + residual
          v += bias[gc];
          ((float*)Out)[oi] = resid[(size_t)gr * LDO + gc] + gamma[gc] * v;
        } else if constexpr (MODE == 4) {        // fc1 + bias + exact GELU
          v += bias[gc];
          v = 0.5f * v * (1.0f + erff(v * 0.70710678118654752f));
          ((__bf16*)Out)[oi] = f2bf(v);
        } else {                                 // fc2 + gamma2 + residual
          v += bias[gc];
          ((float*)Out)[oi] = resid[(size_t)gr * LDO + gc] + gamma[gc] * v;
        }
      }
    }
  }
}

// --------------------------- LayerNorm (f32 -> bf16) -----------------------
__global__ __launch_bounds__(256) void ln_kernel(
    const float* __restrict__ x, const float* __restrict__ g,
    const float* __restrict__ b, __bf16* __restrict__ out)
{
  const int row = blockIdx.x;
  const float* xr = x + (size_t)row * CD;
  const int t = threadIdx.x;
  float v[4], s = 0.0f, s2 = 0.0f;
  for (int i = 0; i < 4; ++i) {
    v[i] = xr[t + i * 256];
    s += v[i]; s2 += v[i] * v[i];
  }
  for (int m = 16; m >= 1; m >>= 1) {
    s  += __shfl_xor(s,  m, 32);
    s2 += __shfl_xor(s2, m, 32);
  }
  __shared__ float ss[8], ss2[8];
  const int wid = t >> 5, lane = t & 31;
  if (lane == 0) { ss[wid] = s; ss2[wid] = s2; }
  __syncthreads();
  float ts = 0.0f, ts2 = 0.0f;
  for (int i = 0; i < 8; ++i) { ts += ss[i]; ts2 += ss2[i]; }
  const float mean = ts * (1.0f / CD);
  const float var  = ts2 * (1.0f / CD) - mean * mean;
  const float rstd = rsqrtf(var + LN_EPS);
  for (int i = 0; i < 4; ++i) {
    const int c = t + i * 256;
    out[(size_t)row * CD + c] = f2bf((v[i] - mean) * rstd * g[c] + b[c]);
  }
}

// --------------------------- Softmax (in-place, f32) -----------------------
__global__ __launch_bounds__(256) void softmax_kernel(float* __restrict__ attn)
{
  float* p = attn + (size_t)blockIdx.x * NSEQ;
  const int t = threadIdx.x;
  float v[4], mx = -3.402823466e38f;
  for (int i = 0; i < 4; ++i) { v[i] = p[t + i * 256]; mx = fmaxf(mx, v[i]); }
  for (int m = 16; m >= 1; m >>= 1) mx = fmaxf(mx, __shfl_xor(mx, m, 32));
  __shared__ float sm[8], sw[8];
  const int wid = t >> 5, lane = t & 31;
  if (lane == 0) sm[wid] = mx;
  __syncthreads();
  float gmx = sm[0];
  for (int i = 1; i < 8; ++i) gmx = fmaxf(gmx, sm[i]);
  float s = 0.0f;
  for (int i = 0; i < 4; ++i) { v[i] = __expf(v[i] - gmx); s += v[i]; }
  for (int m = 16; m >= 1; m >>= 1) s += __shfl_xor(s, m, 32);
  if (lane == 0) sw[wid] = s;
  __syncthreads();
  float gs = 0.0f;
  for (int i = 0; i < 8; ++i) gs += sw[i];
  const float inv = 1.0f / gs;
  for (int i = 0; i < 4; ++i) p[t + i * 256] = v[i] * inv;
}

// --------------------------- Helpers ---------------------------------------
__global__ __launch_bounds__(256) void cvt_bf16_kernel(
    const float* __restrict__ in, __bf16* __restrict__ out, int n)
{
  const int i = blockIdx.x * 256 + threadIdx.x;
  if (i < n) out[i] = f2bf(in[i]);
}

// vT[z][d][m] = qkv[b*N + m][2C + h*64 + d]   (z = b*16 + h)
__global__ __launch_bounds__(256) void transpose_v_kernel(
    const __bf16* __restrict__ qkv, __bf16* __restrict__ vT)
{
  const int i = blockIdx.x * 256 + threadIdx.x;
  const int m  = i & (NSEQ - 1);
  const int r  = i >> 10;
  const int d  = r & (HDID - 1);
  const int zz = r >> 6;
  const int b  = zz >> 4, h = zz & 15;
  vT[i] = qkv[(size_t)(b * NSEQ + m) * 3072 + 2 * CD + h * HDID + d];
}

// ---------------------------------------------------------------------------
extern "C" void kernel_launch(void* const* d_in, const int* in_sizes, int n_in,
                              void* d_out, int out_size, void* d_ws, size_t ws_size,
                              hipStream_t stream)
{
  (void)in_sizes; (void)n_in; (void)out_size; (void)ws_size;
  const float* x      = (const float*)d_in[0];
  const float* ln1_g  = (const float*)d_in[1];
  const float* ln1_b  = (const float*)d_in[2];
  const float* qkv_w  = (const float*)d_in[3];
  const float* proj_w = (const float*)d_in[4];
  const float* proj_b = (const float*)d_in[5];
  const float* gamma1 = (const float*)d_in[6];
  const float* ln2_g  = (const float*)d_in[7];
  const float* ln2_b  = (const float*)d_in[8];
  const float* fc1_w  = (const float*)d_in[9];
  const float* fc1_b  = (const float*)d_in[10];
  const float* fc2_w  = (const float*)d_in[11];
  const float* fc2_b  = (const float*)d_in[12];
  const float* gamma2 = (const float*)d_in[13];

  float* out_x    = (float*)d_out;
  float* out_attn = out_x + (size_t)MROWS * CD;

  char* w = (char*)d_ws;
  auto take = [&](size_t bytes) -> char* {
    char* p = w; w += (bytes + 255) & ~(size_t)255; return p;
  };
  __bf16* n1    = (__bf16*)take((size_t)MROWS * CD * 2);
  __bf16* qkv   = (__bf16*)take((size_t)MROWS * 3072 * 2);
  __bf16* vT    = (__bf16*)take((size_t)BSZ * HEADS * HDID * NSEQ * 2);
  __bf16* ao    = (__bf16*)take((size_t)MROWS * CD * 2);
  float*  x1    = (float*) take((size_t)MROWS * CD * 4);
  __bf16* n2    = (__bf16*)take((size_t)MROWS * CD * 2);
  __bf16* hbuf  = (__bf16*)take((size_t)MROWS * HID * 2);
  __bf16* wqkv  = (__bf16*)take((size_t)3 * CD * CD * 2);
  __bf16* wproj = (__bf16*)take((size_t)CD * CD * 2);
  __bf16* wfc1  = (__bf16*)take((size_t)HID * CD * 2);
  __bf16* wfc2  = (__bf16*)take((size_t)CD * HID * 2);

  // 0) weights -> bf16
  cvt_bf16_kernel<<<(3 * CD * CD) / 256, 256, 0, stream>>>(qkv_w,  wqkv,  3 * CD * CD);
  cvt_bf16_kernel<<<(CD * CD) / 256,     256, 0, stream>>>(proj_w, wproj, CD * CD);
  cvt_bf16_kernel<<<(HID * CD) / 256,    256, 0, stream>>>(fc1_w,  wfc1,  HID * CD);
  cvt_bf16_kernel<<<(CD * HID) / 256,    256, 0, stream>>>(fc2_w,  wfc2,  CD * HID);

  // 1) LN1
  ln_kernel<<<MROWS, 256, 0, stream>>>(x, ln1_g, ln1_b, n1);

  // 2) QKV GEMM: [4096,1024] x [3072,1024]^T -> bf16 qkv (q scaled)
  gemm_kernel<0><<<dim3(3072 / 128, MROWS / 128, 1), 256, 0, stream>>>(
      n1, wqkv, qkv, nullptr, nullptr, nullptr);

  // 3) v transpose for AV GEMM
  transpose_v_kernel<<<(BSZ * HEADS * HDID * NSEQ) / 256, 256, 0, stream>>>(qkv, vT);

  // 4) scores: per (b,h)  q[1024,64] x k[1024,64]^T -> f32 attn in d_out
  gemm_kernel<1><<<dim3(NSEQ / 128, NSEQ / 128, BSZ * HEADS), 256, 0, stream>>>(
      qkv, qkv, out_attn, nullptr, nullptr, nullptr);

  // 5) softmax in place on d_out attn region
  softmax_kernel<<<BSZ * HEADS * NSEQ, 256, 0, stream>>>(out_attn);

  // 6) AV: per (b,h)  attn[1024,1024](f32) x vT[64,1024]^T -> bf16 ao
  gemm_kernel<2><<<dim3(1, NSEQ / 128, BSZ * HEADS), 256, 0, stream>>>(
      out_attn, vT, ao, nullptr, nullptr, nullptr);

  // 7) proj + gamma1 + residual -> f32 x1
  gemm_kernel<3><<<dim3(CD / 128, MROWS / 128, 1), 256, 0, stream>>>(
      ao, wproj, x1, proj_b, x, gamma1);

  // 8) LN2
  ln_kernel<<<MROWS, 256, 0, stream>>>(x1, ln2_g, ln2_b, n2);

  // 9) FC1 + GELU -> bf16 h
  gemm_kernel<4><<<dim3(HID / 128, MROWS / 128, 1), 256, 0, stream>>>(
      n2, wfc1, hbuf, fc1_b, nullptr, nullptr);

  // 10) FC2 + gamma2 + residual -> f32 d_out (x part)
  gemm_kernel<5><<<dim3(CD / 128, MROWS / 128, 1), 256, 0, stream>>>(
      hbuf, wfc2, out_x, fc2_b, x1, gamma2);
}
